// CTMv2Block_1322849927850
// MI455X (gfx1250) — compile-verified, matrix operand order checked by reading, not run on hardware
//
#include <hip/hip_runtime.h>

#define BT    1024
#define DD    1024
#define HH    16
#define HD    64
#define TT    512
#define BB    2
#define NSY   512
#define GG    256
#define NIN   204
#define NAT   307
#define NOUTN 307
#define NMO   206
#define OFF_IN  0
#define OFF_AT  204
#define OFF_OUT 511
#define OFF_MO  818
#define CATP  1600   // padded bf16 concat stride (multiple of 32)

typedef __attribute__((ext_vector_type(16))) __bf16 v16bf;
typedef __attribute__((ext_vector_type(8)))  float  v8f;

static __device__ inline v8f wmma_bf16(v16bf a, v16bf b, v8f c) {
  return __builtin_amdgcn_wmma_f32_16x16x32_bf16(false, a, false, b, (short)0, c,
                                                 false, false);
}
static __device__ inline float sigm(float x) { return 1.f / (1.f + __expf(-x)); }

// ---------------------------------------------------------------------------
// GEMM: C[M,N] (fp32) = A[M,K] (bf16 row-major) x Bt[N,K] (bf16 transposed)
// Requirements (guaranteed by the host-side buffer layout):
//   M % 64 == 0, K % 32 == 0, lda/ldb % 8 == 0,
//   A zero-padded through K, Bt zero-padded through K and through pad128(N).
// Block tile 64x128, 8 waves, wave tile 16x64. Double-buffered LDS staged
// with global_load_async_to_lds_b128 (ASYNCcnt): DMA of tile k+1 overlaps
// WMMA on tile k. No conditional loads anywhere; only the C store is guarded.
// ---------------------------------------------------------------------------
__global__ __launch_bounds__(256) void k_gemm(const __bf16* __restrict__ A,
                                              const __bf16* __restrict__ Bt,
                                              float* __restrict__ C,
                                              int N, int K,
                                              int lda, int ldb, int ldc) {
  __shared__ __bf16 Ah[2][64][32];   // [buf][m][k]
  __shared__ __bf16 Bh[2][128][32];  // [buf][n][k]
  const int tid = threadIdx.x;
  const int lane = tid & 31, wave = tid >> 5;
  const int wm = wave >> 1, wn = wave & 1;
  const int lr = lane & 15, lh = lane >> 4;
  const int bm = blockIdx.y * 64, bn = blockIdx.x * 128;
  v8f acc[4] = {};

  auto stage_async = [&](int k0, int p) {
    {  // A tile 64x32 bf16 = 256 16B chunks, one per thread
      int r = tid >> 2, c8 = (tid & 3) * 8;
      unsigned l = (unsigned)(uintptr_t)&Ah[p][r][c8];
      unsigned long long g =
          (unsigned long long)(uintptr_t)&A[(size_t)(bm + r) * lda + k0 + c8];
      asm volatile("global_load_async_to_lds_b128 %0, %1, off" ::"v"(l), "v"(g)
                   : "memory");
    }
#pragma unroll
    for (int i = 0; i < 2; ++i) {  // B tile 128x32 bf16 = 512 16B chunks
      int idx = tid + i * 256;
      int n = idx >> 2, c8 = (idx & 3) * 8;
      unsigned l = (unsigned)(uintptr_t)&Bh[p][n][c8];
      unsigned long long g =
          (unsigned long long)(uintptr_t)&Bt[(size_t)(bn + n) * ldb + k0 + c8];
      asm volatile("global_load_async_to_lds_b128 %0, %1, off" ::"v"(l), "v"(g)
                   : "memory");
    }
  };

  stage_async(0, 0);
  asm volatile("s_wait_asynccnt 0" ::: "memory");
  __syncthreads();
  for (int k0 = 0; k0 < K; k0 += 32) {
    const int p = (k0 >> 5) & 1;
    if (k0 + 32 < K) stage_async(k0 + 32, p ^ 1);  // DMA next tile
    // preload all fragments, then burst the 4 WMMAs
    v16bf a  = *(const v16bf*)&Ah[p][wm * 16 + lr][lh * 16];
    v16bf b0 = *(const v16bf*)&Bh[p][wn * 64 +  0 + lr][lh * 16];
    v16bf b1 = *(const v16bf*)&Bh[p][wn * 64 + 16 + lr][lh * 16];
    v16bf b2 = *(const v16bf*)&Bh[p][wn * 64 + 32 + lr][lh * 16];
    v16bf b3 = *(const v16bf*)&Bh[p][wn * 64 + 48 + lr][lh * 16];
    acc[0] = wmma_bf16(a, b0, acc[0]);
    acc[1] = wmma_bf16(a, b1, acc[1]);
    acc[2] = wmma_bf16(a, b2, acc[2]);
    acc[3] = wmma_bf16(a, b3, acc[3]);
    asm volatile("s_wait_asynccnt 0" ::: "memory");
    __syncthreads();
  }
#pragma unroll
  for (int nt = 0; nt < 4; ++nt)
#pragma unroll
    for (int r = 0; r < 8; ++r) {
      int crow = bm + wm * 16 + r + lh * 8;
      int ccol = bn + wn * 64 + nt * 16 + lr;
      if (ccol < N) C[(size_t)crow * ldc + ccol] = acc[nt][r];
    }
}

// ---------------------------------------------------------------------------
// Flash attention per (b,h): causal softmax(QK^T/8) @ V, fused +tick_embed.
// ---------------------------------------------------------------------------
__global__ __launch_bounds__(128) void k_attn(const float* __restrict__ q,
                                              const float* __restrict__ kmat,
                                              const float* __restrict__ vmat,
                                              float* __restrict__ attn,
                                              const float* __restrict__ tick) {
  __shared__ __bf16 Qs[64][64];
  __shared__ __bf16 Ks[64][64];
  __shared__ __bf16 Vt[64][64];      // transposed: [d][k]
  __shared__ __bf16 Ps[4][16][64];   // per-wave P staging
  const int tid = threadIdx.x, lane = tid & 31, wave = tid >> 5;
  const int lr = lane & 15, lh = lane >> 4;
  const int bh = blockIdx.y, b = bh >> 4, h = bh & 15;
  const int qb = blockIdx.x * 64;
#pragma unroll 8
  for (int i = 0; i < 32; ++i) {
    int idx = tid + i * 128;
    int r = idx >> 6, d = idx & 63;
    Qs[r][d] = (__bf16)(q[(size_t)(b * TT + qb + r) * DD + h * HD + d] * 0.125f);
  }
  v8f o[4] = {};
  float m8[8], l8[8];
#pragma unroll
  for (int r = 0; r < 8; ++r) { m8[r] = -1e30f; l8[r] = 0.f; }
  for (int j = 0; j <= (int)blockIdx.x; ++j) {
    const int kb = j * 64;
    __syncthreads();
#pragma unroll 8
    for (int i = 0; i < 32; ++i) {
      int idx = tid + i * 128;
      int r = idx >> 6, d = idx & 63;
      size_t base = (size_t)(b * TT + kb + r) * DD + h * HD + d;
      Ks[r][d] = (__bf16)kmat[base];
      Vt[d][r] = (__bf16)vmat[base];
    }
    __syncthreads();
    v8f s[4] = {};
#pragma unroll
    for (int kh = 0; kh < 2; ++kh) {
      v16bf a  = *(const v16bf*)&Qs[wave * 16 + lr][kh * 32 + lh * 16];
      v16bf b0 = *(const v16bf*)&Ks[ 0 + lr][kh * 32 + lh * 16];
      v16bf b1 = *(const v16bf*)&Ks[16 + lr][kh * 32 + lh * 16];
      v16bf b2 = *(const v16bf*)&Ks[32 + lr][kh * 32 + lh * 16];
      v16bf b3 = *(const v16bf*)&Ks[48 + lr][kh * 32 + lh * 16];
      s[0] = wmma_bf16(a, b0, s[0]);
      s[1] = wmma_bf16(a, b1, s[1]);
      s[2] = wmma_bf16(a, b2, s[2]);
      s[3] = wmma_bf16(a, b3, s[3]);
    }
    const bool diag = (kb == qb);
#pragma unroll
    for (int r = 0; r < 8; ++r) {
      const int qrow = wave * 16 + r + lh * 8;
      float mx = -1e30f;
#pragma unroll
      for (int nt = 0; nt < 4; ++nt) {
        float sv = s[nt][r];
        if (diag && (nt * 16 + lr > qrow)) sv = -1e30f;
        s[nt][r] = sv;
        mx = fmaxf(mx, sv);
      }
#pragma unroll
      for (int msk = 8; msk >= 1; msk >>= 1) mx = fmaxf(mx, __shfl_xor(mx, msk, 32));
      float mnew = fmaxf(m8[r], mx);
      float corr = __expf(m8[r] - mnew);
      float rs = 0.f;
#pragma unroll
      for (int nt = 0; nt < 4; ++nt) {
        float p = __expf(s[nt][r] - mnew);
        s[nt][r] = p;
        rs += p;
      }
#pragma unroll
      for (int msk = 8; msk >= 1; msk >>= 1) rs += __shfl_xor(rs, msk, 32);
      l8[r] = l8[r] * corr + rs;
      m8[r] = mnew;
#pragma unroll
      for (int nt = 0; nt < 4; ++nt) o[nt][r] *= corr;
    }
#pragma unroll
    for (int r = 0; r < 8; ++r)
#pragma unroll
      for (int nt = 0; nt < 4; ++nt)
        Ps[wave][r + lh * 8][nt * 16 + lr] = (__bf16)s[nt][r];
    __syncthreads();
#pragma unroll
    for (int kh = 0; kh < 2; ++kh) {
      v16bf a  = *(const v16bf*)&Ps[wave][lr][kh * 32 + lh * 16];
      v16bf b0 = *(const v16bf*)&Vt[ 0 + lr][kh * 32 + lh * 16];
      v16bf b1 = *(const v16bf*)&Vt[16 + lr][kh * 32 + lh * 16];
      v16bf b2 = *(const v16bf*)&Vt[32 + lr][kh * 32 + lh * 16];
      v16bf b3 = *(const v16bf*)&Vt[48 + lr][kh * 32 + lh * 16];
      o[0] = wmma_bf16(a, b0, o[0]);
      o[1] = wmma_bf16(a, b1, o[1]);
      o[2] = wmma_bf16(a, b2, o[2]);
      o[3] = wmma_bf16(a, b3, o[3]);
    }
  }
#pragma unroll
  for (int r = 0; r < 8; ++r) {
    float inv = 1.f / l8[r];
    size_t base = (size_t)(b * TT + qb + wave * 16 + r + lh * 8) * DD + h * HD;
#pragma unroll
    for (int nt = 0; nt < 4; ++nt)
      attn[base + nt * 16 + lr] = o[nt][r] * inv + tick[h * HD + nt * 16 + lr];
  }
}

// --------------------------- conversion kernels ----------------------------
__global__ void k_f2bf(const float* src, __bf16* dst, int total) {
  int i = blockIdx.x * blockDim.x + threadIdx.x;
  if (i >= total) return;
  dst[i] = (__bf16)src[i];
}
// weight W[K,N] fp32 -> transposed bf16 Bt[Nr,padK]; zero pad rows n>=N and
// columns k>=K so the GEMM needs no bounds handling.
__global__ void k_w2bt(const float* W, __bf16* Bt, int K, int N, int padK,
                       int Nr) {
  int i = blockIdx.x * blockDim.x + threadIdx.x;
  if (i >= Nr * padK) return;
  int n = i / padK, k = i % padK;
  Bt[i] = (n < N && k < K) ? (__bf16)W[(size_t)k * N + n] : (__bf16)0.f;
}

// --------------------------- elementwise kernels ---------------------------
__global__ void k_rmshead(float* buf) {  // rms over contiguous groups of 64
  int g = blockIdx.x;
  size_t base = (size_t)g * 64;
  float v = buf[base + threadIdx.x];
  __shared__ float red[64];
  red[threadIdx.x] = v * v;
  __syncthreads();
  for (int st = 32; st > 0; st >>= 1) {
    if ((int)threadIdx.x < st) red[threadIdx.x] += red[threadIdx.x + st];
    __syncthreads();
  }
  buf[base + threadIdx.x] = v * rsqrtf(red[0] / 64.f + 1e-6f);
}

__global__ void k_state_init(float* state, const float* s0i, const float* s0a,
                             const float* s0o, const float* s0m) {
  int i = blockIdx.x * blockDim.x + threadIdx.x;
  if (i >= BT * DD) return;
  int d = i % DD;
  float v;
  if (d < OFF_AT) v = s0i[d];
  else if (d < OFF_OUT) v = s0a[d - OFF_AT];
  else if (d < OFF_MO) v = s0o[d - OFF_OUT];
  else v = s0m[d - OFF_MO];
  state[i] = v;
}

__global__ void k_bcast(float* dst, const float* src, int per, int total) {
  int i = blockIdx.x * blockDim.x + threadIdx.x;
  if (i >= total) return;
  dst[i] = src[i % per];
}

__global__ void k_init_ab(float* alpha, float* beta, const float* state,
                          const int* il, const int* ir) {
  int i = blockIdx.x * blockDim.x + threadIdx.x;
  if (i >= BT * NSY) return;
  int bt = i / NSY, j = i % NSY;
  alpha[i] = state[(size_t)bt * DD + il[j]] * state[(size_t)bt * DD + ir[j]];
  beta[i] = 1.f;
}

__global__ void k_update_ab(float* alpha, float* beta, const float* state,
                            const int* il, const int* ir, const float* dec) {
  int i = blockIdx.x * blockDim.x + threadIdx.x;
  if (i >= BT * NSY) return;
  int bt = i / NSY, j = i % NSY;
  float d = __expf(-fminf(fmaxf(dec[j], 0.f), 15.f));
  alpha[i] = d * alpha[i] +
             state[(size_t)bt * DD + il[j]] * state[(size_t)bt * DD + ir[j]];
  beta[i] = d * beta[i] + 1.f;
}

// sync = alpha * rsqrt(beta), emitted directly as bf16 GEMM operand
__global__ void k_syncz(__bf16* s, const float* a, const float* b) {
  int i = blockIdx.x * blockDim.x + threadIdx.x;
  if (i >= BT * NSY) return;
  s[i] = (__bf16)(a[i] * rsqrtf(b[i]));
}

__global__ void k_glu(const float* z, float* outp, int N) {
  int i = blockIdx.x * blockDim.x + threadIdx.x;
  if (i >= BT * N) return;
  int bt = i / N, c = i % N;
  float a = z[(size_t)bt * 2 * N + c];
  float b = z[(size_t)bt * 2 * N + N + c];
  outp[i] = a * sigm(b);
}

__global__ __launch_bounds__(256) void k_gluln(const float* z, float* pre, int N,
                                               const float* g, const float* bv) {
  int bt = blockIdx.x;
  __shared__ float r1[256], r2[256];
  float s = 0.f, s2 = 0.f;
  for (int c = threadIdx.x; c < N; c += 256) {
    float a = z[(size_t)bt * 2 * N + c];
    float b = z[(size_t)bt * 2 * N + N + c];
    float gv = a * sigm(b);
    pre[(size_t)bt * N + c] = gv;
    s += gv;
    s2 += gv * gv;
  }
  r1[threadIdx.x] = s;
  r2[threadIdx.x] = s2;
  __syncthreads();
  for (int st = 128; st > 0; st >>= 1) {
    if ((int)threadIdx.x < st) {
      r1[threadIdx.x] += r1[threadIdx.x + st];
      r2[threadIdx.x] += r2[threadIdx.x + st];
    }
    __syncthreads();
  }
  float mu = r1[0] / N;
  float var = r2[0] / N - mu * mu;
  float inv = rsqrtf(var + 1e-5f);
  for (int c = threadIdx.x; c < N; c += 256) {
    float gv = pre[(size_t)bt * N + c];
    pre[(size_t)bt * N + c] = (gv - mu) * inv * g[c] + bv[c];
  }
}

// concat up to 3 fp32 sources into padded bf16 GEMM operand (stride CATP);
// zero-fills [W, Wp) so the GEMM can run with K = Wp (multiple of 32).
__global__ void k_cat3(__bf16* dst, int W, int Wp, const float* p0, int st0,
                       int w0, const float* p1, int st1, int w1, const float* p2,
                       int st2, int w2) {
  int i = blockIdx.x * blockDim.x + threadIdx.x;
  if (i >= BT * Wp) return;
  int bt = i / Wp, c = i % Wp;
  float v = 0.f;
  if (c < w0) v = p0[(size_t)bt * st0 + c];
  else if (c < w0 + w1) v = p1[(size_t)bt * st1 + (c - w0)];
  else if (c < W) v = p2[(size_t)bt * st2 + (c - w0 - w1)];
  dst[(size_t)bt * CATP + c] = (__bf16)v;
}

__global__ void k_shift(float* tr, const float* pre, int N, int M) {
  int i = blockIdx.x * blockDim.x + threadIdx.x;
  if (i >= BT * N) return;
  float* p = tr + (size_t)i * M;
  for (int m = 0; m < M - 1; ++m) p[m] = p[m + 1];
  p[M - 1] = pre[i];
}

// NLM stage, 2*16 outputs -> GLU -> 16 (einsum bnm,mon->bno then GLU)
__global__ void k_nlm16(const float* in, int Min, const float* w, const float* bias,
                        int N, float* out) {
  int i = blockIdx.x * blockDim.x + threadIdx.x;
  if (i >= BT * N) return;
  int n = i % N;
  const float* ip = in + (size_t)i * Min;
  float acc[32];
#pragma unroll
  for (int oo = 0; oo < 32; ++oo) acc[oo] = bias[n * 32 + oo];
  for (int m = 0; m < Min; ++m) {
    float tm = ip[m];
    const float* wr = w + ((size_t)m * 32) * N + n;
#pragma unroll
    for (int oo = 0; oo < 32; ++oo) acc[oo] += tm * wr[(size_t)oo * N];
  }
  float* op = out + (size_t)i * 16;
#pragma unroll
  for (int oh = 0; oh < 16; ++oh) op[oh] = acc[oh] * sigm(acc[16 + oh]);
}

// NLM stage, 2 outputs -> GLU -> scalar, written into a state slice
__global__ void k_nlm1(const float* in, int Min, const float* w, const float* bias,
                       int N, float* state, int off) {
  int i = blockIdx.x * blockDim.x + threadIdx.x;
  if (i >= BT * N) return;
  int bt = i / N, n = i % N;
  const float* ip = in + (size_t)i * Min;
  float z0 = bias[n * 2 + 0], z1 = bias[n * 2 + 1];
  for (int m = 0; m < Min; ++m) {
    float tm = ip[m];
    z0 += tm * w[((size_t)m * 2 + 0) * N + n];
    z1 += tm * w[((size_t)m * 2 + 1) * N + n];
  }
  state[(size_t)bt * DD + off + n] = z0 * sigm(z1);
}

// ------------------------------- host side --------------------------------
static inline void gemm(const __bf16* A, const __bf16* Bt, float* C, int M, int N,
                        int K, int lda, int ldb, int ldc, hipStream_t s) {
  dim3 grid((N + 127) / 128, M / 64);
  k_gemm<<<grid, 256, 0, s>>>(A, Bt, C, N, K, lda, ldb, ldc);
}
#define GRID1(total) dim3(((total) + 255) / 256), dim3(256)
static inline int pad32(int k) { return (k + 31) & ~31; }
static inline int pad128(int n) { return (n + 127) & ~127; }

extern "C" void kernel_launch(void* const* d_in, const int* in_sizes, int n_in,
                              void* d_out, int out_size, void* d_ws, size_t ws_size,
                              hipStream_t stream) {
  (void)in_sizes; (void)n_in; (void)out_size; (void)ws_size;
  const float* x       = (const float*)d_in[0];
  const float* Wq      = (const float*)d_in[1];
  const float* Wk      = (const float*)d_in[2];
  const float* Wv      = (const float*)d_in[3];
  const float* Wg      = (const float*)d_in[4];
  const float* W_in    = (const float*)d_in[5];
  const float* ln_in_g = (const float*)d_in[6];
  const float* ln_in_b = (const float*)d_in[7];
  const float* W_at    = (const float*)d_in[8];
  const float* ln_at_g = (const float*)d_in[9];
  const float* ln_at_b = (const float*)d_in[10];
  const float* W_out   = (const float*)d_in[11];
  const float* ln_out_g= (const float*)d_in[12];
  const float* ln_out_b= (const float*)d_in[13];
  const float* W_mo    = (const float*)d_in[14];
  const float* ln_mo_g = (const float*)d_in[15];
  const float* ln_mo_b = (const float*)d_in[16];
  const float* tick    = (const float*)d_in[17];
  const float* dec_out = (const float*)d_in[18];
  const float* dec_act = (const float*)d_in[19];
  const float* Wc      = (const float*)d_in[20];
  const float* in_w1   = (const float*)d_in[21];
  const float* in_b1   = (const float*)d_in[22];
  const float* in_s0   = (const float*)d_in[23];
  const float* in_tr0  = (const float*)d_in[24];
  const float* at_w1   = (const float*)d_in[25];
  const float* at_b1   = (const float*)d_in[26];
  const float* at_w2   = (const float*)d_in[27];
  const float* at_b2   = (const float*)d_in[28];
  const float* at_s0   = (const float*)d_in[29];
  const float* at_tr0  = (const float*)d_in[30];
  const float* out_w1  = (const float*)d_in[31];
  const float* out_b1  = (const float*)d_in[32];
  const float* out_w2  = (const float*)d_in[33];
  const float* out_b2  = (const float*)d_in[34];
  const float* out_s0  = (const float*)d_in[35];
  const float* out_tr0 = (const float*)d_in[36];
  const float* mo_w1   = (const float*)d_in[37];
  const float* mo_b1   = (const float*)d_in[38];
  const float* mo_s0   = (const float*)d_in[39];
  const float* mo_tr0  = (const float*)d_in[40];
  const int* idx_out_l = (const int*)d_in[41];
  const int* idx_out_r = (const int*)d_in[42];
  const int* idx_act_l = (const int*)d_in[43];
  const int* idx_act_r = (const int*)d_in[44];
  float* outp = (float*)d_out;

  // byte allocator over d_ws, 32B-aligned slices
  char* base = (char*)d_ws;
  size_t off = 0;
  auto allocf = [&](size_t n) {
    float* p = (float*)(base + off);
    off = (off + n * 4 + 31) & ~(size_t)31;
    return p;
  };
  auto allocb = [&](size_t n) {
    __bf16* p = (__bf16*)(base + off);
    off = (off + n * 2 + 31) & ~(size_t)31;
    return p;
  };
  // fp32 buffers
  float* kk     = allocf((size_t)BT * DD);
  float* vv     = allocf((size_t)BT * DD);
  float* state  = allocf((size_t)BT * DD);
  float* tr_in  = allocf((size_t)BT * NIN * 4);
  float* tr_at  = allocf((size_t)BT * NAT * 128);
  float* tr_out = allocf((size_t)BT * NOUTN * 16);
  float* tr_mo  = allocf((size_t)BT * NMO * 4);
  float* a_out  = allocf((size_t)BT * NSY);
  float* b_out  = allocf((size_t)BT * NSY);
  float* a_act  = allocf((size_t)BT * NSY);
  float* b_act  = allocf((size_t)BT * NSY);
  float* qbuf   = allocf((size_t)BT * DD);
  float* attnb  = allocf((size_t)BT * DD);
  float* gz     = allocf((size_t)BT * 2 * GG);
  float* gfeat  = allocf((size_t)BT * GG);
  float* zb     = allocf((size_t)BT * 640);
  float* preb   = allocf((size_t)BT * 320);
  float* hid    = allocf((size_t)BT * NAT * 16);
  // bf16 GEMM operands
  __bf16* xb       = allocb((size_t)BT * DD);
  __bf16* syncb    = allocb((size_t)BT * NSY);
  __bf16* state_bf = allocb((size_t)BT * DD);
  __bf16* cat_bf   = allocb((size_t)BT * CATP);
  const int pk_in = pad32(1486), pk_at = pad32(1228), pk_out = pad32(1587),
            pk_mo = pad32(563);
  const int nr_in = pad128(2 * NIN), nr_at = pad128(2 * NAT),
            nr_out = pad128(2 * NOUTN), nr_mo = pad128(2 * NMO);
  __bf16* WkT  = allocb((size_t)DD * DD);
  __bf16* WvT  = allocb((size_t)DD * DD);
  __bf16* WqT  = allocb((size_t)DD * NSY);
  __bf16* WgT  = allocb((size_t)2 * GG * DD);
  __bf16* WinT = allocb((size_t)nr_in * pk_in);
  __bf16* WatT = allocb((size_t)nr_at * pk_at);
  __bf16* WouT = allocb((size_t)nr_out * pk_out);
  __bf16* WmoT = allocb((size_t)nr_mo * pk_mo);
  __bf16* WcT  = allocb((size_t)DD * NSY);

  // ---- one-time (per launch) weight transpose+convert to bf16 ----
  k_w2bt<<<GRID1(DD * DD), 0, stream>>>(Wk, WkT, DD, DD, DD, DD);
  k_w2bt<<<GRID1(DD * DD), 0, stream>>>(Wv, WvT, DD, DD, DD, DD);
  k_w2bt<<<GRID1(DD * NSY), 0, stream>>>(Wq, WqT, NSY, DD, NSY, DD);
  k_w2bt<<<GRID1(2 * GG * DD), 0, stream>>>(Wg, WgT, DD, 2 * GG, DD, 2 * GG);
  k_w2bt<<<GRID1(nr_in * pk_in), 0, stream>>>(W_in, WinT, 1486, 2 * NIN, pk_in, nr_in);
  k_w2bt<<<GRID1(nr_at * pk_at), 0, stream>>>(W_at, WatT, 1228, 2 * NAT, pk_at, nr_at);
  k_w2bt<<<GRID1(nr_out * pk_out), 0, stream>>>(W_out, WouT, 1587, 2 * NOUTN, pk_out, nr_out);
  k_w2bt<<<GRID1(nr_mo * pk_mo), 0, stream>>>(W_mo, WmoT, 563, 2 * NMO, pk_mo, nr_mo);
  k_w2bt<<<GRID1(DD * NSY), 0, stream>>>(Wc, WcT, NSY, DD, NSY, DD);
  k_f2bf<<<GRID1(BT * DD), 0, stream>>>(x, xb, BT * DD);

  // K/V projections (once), per-head RMS on K
  gemm(xb, WkT, kk, BT, DD, DD, DD, DD, DD, stream);
  k_rmshead<<<BT * HH, 64, 0, stream>>>(kk);
  gemm(xb, WvT, vv, BT, DD, DD, DD, DD, DD, stream);

  // init state, traces, sync accumulators
  k_state_init<<<GRID1(BT * DD), 0, stream>>>(state, in_s0, at_s0, out_s0, mo_s0);
  k_bcast<<<GRID1(BT * NIN * 4), 0, stream>>>(tr_in, in_tr0, NIN * 4, BT * NIN * 4);
  k_bcast<<<GRID1(BT * NAT * 128), 0, stream>>>(tr_at, at_tr0, NAT * 128, BT * NAT * 128);
  k_bcast<<<GRID1(BT * NOUTN * 16), 0, stream>>>(tr_out, out_tr0, NOUTN * 16, BT * NOUTN * 16);
  k_bcast<<<GRID1(BT * NMO * 4), 0, stream>>>(tr_mo, mo_tr0, NMO * 4, BT * NMO * 4);
  k_init_ab<<<GRID1(BT * NSY), 0, stream>>>(a_out, b_out, state, idx_out_l, idx_out_r);
  k_init_ab<<<GRID1(BT * NSY), 0, stream>>>(a_act, b_act, state, idx_act_l, idx_act_r);

  for (int t = 0; t < 8; ++t) {
    // sync_act -> Q -> rms -> flash attention (+tick)
    k_syncz<<<GRID1(BT * NSY), 0, stream>>>(syncb, a_act, b_act);
    gemm(syncb, WqT, qbuf, BT, DD, NSY, NSY, NSY, DD, stream);
    k_rmshead<<<BT * HH, 64, 0, stream>>>(qbuf);
    k_attn<<<dim3(TT / 64, BB * HH), 128, 0, stream>>>(qbuf, kk, vv, attnb,
                                                       tick + (size_t)t * DD);
    // gfeat from previous-tick state
    k_f2bf<<<GRID1(BT * DD), 0, stream>>>(state, state_bf, BT * DD);
    gemm(state_bf, WgT, gz, BT, 2 * GG, DD, DD, DD, 2 * GG, stream);
    k_glu<<<GRID1(BT * GG), 0, stream>>>(gz, gfeat, GG);
    if (t < 2) {  // in-branch
      k_cat3<<<GRID1(BT * pk_in), 0, stream>>>(cat_bf, 1486, pk_in, attnb, DD, DD,
                                               state + OFF_MO, DD, NMO, gfeat, GG, GG);
      gemm(cat_bf, WinT, zb, BT, 2 * NIN, pk_in, CATP, pk_in, 2 * NIN, stream);
      k_gluln<<<BT, 256, 0, stream>>>(zb, preb, NIN, ln_in_g, ln_in_b);
      k_shift<<<GRID1(BT * NIN), 0, stream>>>(tr_in, preb, NIN, 4);
      k_nlm1<<<GRID1(BT * NIN), 0, stream>>>(tr_in, 4, in_w1, in_b1, NIN, state, OFF_IN);
    }
    // at-branch
    k_cat3<<<GRID1(BT * pk_at), 0, stream>>>(cat_bf, 1228, pk_at, state + OFF_IN,
                                             DD, NIN, attnb, DD, DD, attnb, DD, 0);
    gemm(cat_bf, WatT, zb, BT, 2 * NAT, pk_at, CATP, pk_at, 2 * NAT, stream);
    k_gluln<<<BT, 256, 0, stream>>>(zb, preb, NAT, ln_at_g, ln_at_b);
    k_shift<<<GRID1(BT * NAT), 0, stream>>>(tr_at, preb, NAT, 128);
    k_nlm16<<<GRID1(BT * NAT), 0, stream>>>(tr_at, 128, at_w1, at_b1, NAT, hid);
    k_nlm1<<<GRID1(BT * NAT), 0, stream>>>(hid, 16, at_w2, at_b2, NAT, state, OFF_AT);
    // out-branch
    k_cat3<<<GRID1(BT * pk_out), 0, stream>>>(cat_bf, 1587, pk_out, state + OFF_AT,
                                              DD, NAT, attnb, DD, DD, gfeat, GG, GG);
    gemm(cat_bf, WouT, zb, BT, 2 * NOUTN, pk_out, CATP, pk_out, 2 * NOUTN, stream);
    k_gluln<<<BT, 256, 0, stream>>>(zb, preb, NOUTN, ln_out_g, ln_out_b);
    k_shift<<<GRID1(BT * NOUTN), 0, stream>>>(tr_out, preb, NOUTN, 16);
    k_nlm16<<<GRID1(BT * NOUTN), 0, stream>>>(tr_out, 16, out_w1, out_b1, NOUTN, hid);
    k_nlm1<<<GRID1(BT * NOUTN), 0, stream>>>(hid, 16, out_w2, out_b2, NOUTN, state, OFF_OUT);
    if (t < 2) {  // mo-branch
      k_cat3<<<GRID1(BT * pk_mo), 0, stream>>>(cat_bf, 563, pk_mo, state + OFF_OUT,
                                               DD, NOUTN, gfeat, GG, GG, gfeat, GG, 0);
      gemm(cat_bf, WmoT, zb, BT, 2 * NMO, pk_mo, CATP, pk_mo, 2 * NMO, stream);
      k_gluln<<<BT, 256, 0, stream>>>(zb, preb, NMO, ln_mo_g, ln_mo_b);
      k_shift<<<GRID1(BT * NMO), 0, stream>>>(tr_mo, preb, NMO, 4);
      k_nlm1<<<GRID1(BT * NMO), 0, stream>>>(tr_mo, 4, mo_w1, mo_b1, NMO, state, OFF_MO);
    }
    // sync-state recurrences
    k_update_ab<<<GRID1(BT * NSY), 0, stream>>>(a_out, b_out, state, idx_out_l,
                                                idx_out_r, dec_out);
    k_update_ab<<<GRID1(BT * NSY), 0, stream>>>(a_act, b_act, state, idx_act_l,
                                                idx_act_r, dec_act);
  }
  // output projection
  k_syncz<<<GRID1(BT * NSY), 0, stream>>>(syncb, a_out, b_out);
  gemm(syncb, WcT, outp, BT, DD, NSY, NSY, NSY, DD, stream);
}
